// RadialLinear_34505767256589
// MI455X (gfx1250) — compile-verified, matrix operand order checked by reading, not run on hardware
//
#include <hip/hip_runtime.h>
#include <hip/hip_bf16.h>

// ---------------------------------------------------------------------------
// RadialLinear (KAN-style RBF layer) for gfx1250 / MI455X
//   y = basis(LN(x)) @ spline_w^T + silu(x) @ base_w^T + base_b
// One fused bf16 WMMA GEMM, K = 1024*8 + 1024 = 9216, double-buffered
// async global->LDS staging (GLOBAL_LOAD_ASYNC_TO_LDS_B128 GVS form).
// ---------------------------------------------------------------------------

#define IN_F   1024
#define OUT_F  1024
#define NROWS  8192
#define NGRID  8
#define KSPL   (IN_F * NGRID)     // 8192
#define KTOT   (KSPL + IN_F)      // 9216
#define DENOM  (4.0f / 7.0f)
#define INV_DENOM (7.0f / 4.0f)
#define LN_EPS 1e-5f

// GEMM tiling
#define KSTEP      64
#define NT         (KTOT / KSTEP)          // 144 (even -> 2x unrolled pipeline)
#define ASTRIDE    144                     // LDS row stride bytes (64 bf16 + 16B pad)
#define TILE_BYTES (128 * ASTRIDE)         // 18432
#define BUF_BYTES  (2 * TILE_BYTES)        // A tile + B tile = 36864
#define LDS_TOTAL  (2 * BUF_BYTES)         // double buffered = 73728

typedef __attribute__((ext_vector_type(16))) __bf16 v16bf;
typedef __attribute__((ext_vector_type(8)))  __bf16 v8bf;
typedef __attribute__((ext_vector_type(8)))  float  v8f;

union Frag16 { v16bf f; v8bf h[2]; };
union Pack8  { __hip_bfloat16 h[8]; int4 v; };
union Pack4  { __hip_bfloat16 h[4]; int2 v; };

// ---- CDNA5 async global->LDS copy, GVS mode: saddr(SGPR64) + voffset(VGPR32)
__device__ __forceinline__ void async_b128_gvs(unsigned lds_byte_off,
                                               unsigned voff,
                                               unsigned long long sbase) {
    asm volatile("global_load_async_to_lds_b128 %0, %1, %2"
                 :: "v"(lds_byte_off), "v"(voff), "s"(sbase) : "memory");
}
#define WAIT_ASYNC(N) asm volatile("s_wait_asynccnt " #N ::: "memory")

// ---------------------------------------------------------------------------
// Kernel 1: per-row LayerNorm + RBF basis (bf16) + silu(x) (bf16)
// ---------------------------------------------------------------------------
__global__ __launch_bounds__(256)
void prep_A_kernel(const float* __restrict__ x,
                   const float* __restrict__ gamma,
                   const float* __restrict__ beta,
                   __hip_bfloat16* __restrict__ Aext) {
    const int n   = blockIdx.x;
    const int tid = threadIdx.x;

    const float4 xv = ((const float4*)(x + (size_t)n * IN_F))[tid];

    float s  = xv.x + xv.y + xv.z + xv.w;
    float s2 = xv.x * xv.x + xv.y * xv.y + xv.z * xv.z + xv.w * xv.w;

    __shared__ float red_s[256];
    __shared__ float red_q[256];
    red_s[tid] = s;
    red_q[tid] = s2;
    __syncthreads();
    #pragma unroll
    for (int off = 128; off > 0; off >>= 1) {
        if (tid < off) {
            red_s[tid] += red_s[tid + off];
            red_q[tid] += red_q[tid + off];
        }
        __syncthreads();
    }
    const float mu  = red_s[0] * (1.0f / IN_F);
    const float var = red_q[0] * (1.0f / IN_F) - mu * mu;
    const float inv = __frsqrt_rn(var + LN_EPS);

    __hip_bfloat16* __restrict__ rowA = Aext + (size_t)n * KTOT;

    const float xs[4] = {xv.x, xv.y, xv.z, xv.w};
    Pack4 silu;
    #pragma unroll
    for (int j = 0; j < 4; ++j) {
        const int i = tid * 4 + j;
        const float xi = xs[j];
        const float xn = (xi - mu) * inv * gamma[i] + beta[i];
        Pack8 p;
        #pragma unroll
        for (int g = 0; g < NGRID; ++g) {
            const float t = (xn - (-2.0f + (float)g * DENOM)) * INV_DENOM;
            p.h[g] = __float2bfloat16(__expf(-t * t));
        }
        *(int4*)(rowA + (size_t)i * NGRID) = p.v;
        silu.h[j] = __float2bfloat16(xi / (1.0f + __expf(-xi)));
    }
    *(int2*)(rowA + KSPL + (size_t)tid * 4) = silu.v;
}

// ---------------------------------------------------------------------------
// Kernel 2: W_ext = concat(bf16(spline_w), bf16(base_w)) rows of length 9216
// ---------------------------------------------------------------------------
__global__ __launch_bounds__(256)
void prep_W_kernel(const float* __restrict__ spline_w,
                   const float* __restrict__ base_w,
                   __hip_bfloat16* __restrict__ Wext) {
    const size_t idx = (size_t)blockIdx.x * 256 + threadIdx.x;
    const int o  = (int)(idx / (KTOT / 4));
    const int k  = (int)(idx % (KTOT / 4)) * 4;
    float4 src;
    if (k < KSPL) src = *(const float4*)(spline_w + (size_t)o * KSPL + k);
    else          src = *(const float4*)(base_w   + (size_t)o * IN_F + (k - KSPL));
    Pack4 p;
    p.h[0] = __float2bfloat16(src.x);
    p.h[1] = __float2bfloat16(src.y);
    p.h[2] = __float2bfloat16(src.z);
    p.h[3] = __float2bfloat16(src.w);
    *(int2*)(Wext + (size_t)o * KTOT + k) = p.v;
}

// ---------------------------------------------------------------------------
// Kernel 3: bf16 WMMA GEMM with double-buffered async LDS staging.
// Block tile 128x128 (8 waves x 64x32), K-step 64, pipeline unrolled x2 so
// buffer offsets are compile-time and K-advance is 2 scalar adds per tile.
// ---------------------------------------------------------------------------
__global__ __launch_bounds__(256)
void gemm_kernel(const __bf16* __restrict__ A,
                 const __bf16* __restrict__ W,
                 const float* __restrict__ bias,
                 float* __restrict__ out) {
    extern __shared__ __align__(16) char smem[];

    const int tid  = threadIdx.x;
    const int lane = tid & 31;
    const int wave = tid >> 5;
    const int wm   = wave & 1;        // 2 M-slots of 64 rows
    const int wn   = wave >> 1;       // 4 N-slots of 32 cols
    const int blockM = blockIdx.x * 128;
    const int blockN = blockIdx.y * 128;

    // ---- async-copy addressing: thread -> 4 x 16B chunks per tile ----------
    const int r0  = tid >> 3;              // rows r0, r0+32, r0+64, r0+96
    const int c16 = (tid & 7) * 16;        // byte offset within 128B K-slice
    const unsigned lds_base = (unsigned)(uintptr_t)(void*)smem;

    unsigned ldsA[4], ldsB[4], voff[4];
    #pragma unroll
    for (int j = 0; j < 4; ++j) {
        ldsA[j] = lds_base + (unsigned)((r0 + 32 * j) * ASTRIDE + c16);
        ldsB[j] = ldsA[j] + TILE_BYTES;
        voff[j] = (unsigned)((r0 + 32 * j) * (KTOT * 2) + c16);   // per-lane, loop-invariant
    }
    // uniform 64-bit bases: advance by 128B per K-tile (scalar adds only)
    unsigned long long sA = (unsigned long long)(uintptr_t)A + (size_t)blockM * (KTOT * 2);
    unsigned long long sB = (unsigned long long)(uintptr_t)W + (size_t)blockN * (KTOT * 2);

    // ---- fragment addressing (CDNA5 16-bit A/B layouts) --------------------
    const int fr   = lane & 15;            // fragment row (A) / col (B)
    const int kq2  = (lane >> 4) * 16;     // lanes 16-31: K starts at +8 elems
    const char* sm = smem;

    v8f acc[4][2];
    #pragma unroll
    for (int mt = 0; mt < 4; ++mt)
        #pragma unroll
        for (int nt = 0; nt < 2; ++nt)
            acc[mt][nt] = (v8f){0.f, 0.f, 0.f, 0.f, 0.f, 0.f, 0.f, 0.f};

    auto issue_tile = [&](unsigned bufofs) {
        #pragma unroll
        for (int j = 0; j < 4; ++j) {
            async_b128_gvs(ldsA[j] + bufofs, voff[j], sA);
            async_b128_gvs(ldsB[j] + bufofs, voff[j], sB);
        }
        sA += KSTEP * 2;
        sB += KSTEP * 2;
    };

    auto compute_tile = [&](unsigned bufofs) {
        const char* sAt = sm + bufofs;
        const char* sBt = sAt + TILE_BYTES;
        #pragma unroll
        for (int ks = 0; ks < KSTEP; ks += 32) {
            Frag16 a[4], b[2];
            #pragma unroll
            for (int mt = 0; mt < 4; ++mt) {
                const char* p = sAt + (wm * 64 + mt * 16 + fr) * ASTRIDE + ks * 2 + kq2;
                a[mt].h[0] = *(const v8bf*)(p);        // K = ks+kq+0..7
                a[mt].h[1] = *(const v8bf*)(p + 32);   // K = ks+kq+16..23
            }
            #pragma unroll
            for (int nt = 0; nt < 2; ++nt) {
                const char* p = sBt + (wn * 32 + nt * 16 + fr) * ASTRIDE + ks * 2 + kq2;
                b[nt].h[0] = *(const v8bf*)(p);
                b[nt].h[1] = *(const v8bf*)(p + 32);
            }
            #pragma unroll
            for (int mt = 0; mt < 4; ++mt)
                #pragma unroll
                for (int nt = 0; nt < 2; ++nt)
                    acc[mt][nt] = __builtin_amdgcn_wmma_f32_16x16x32_bf16(
                        false, a[mt].f, false, b[nt].f,
                        (short)0, acc[mt][nt], false, false);
        }
    };

    // ---- software pipeline: tile kt -> buf0, tile kt+1 -> buf1 (NT even) ---
    issue_tile(0);                         // tile 0 -> buf0
    for (int kt = 0; kt < NT; kt += 2) {
        issue_tile(BUF_BYTES);             // tile kt+1 -> buf1
        WAIT_ASYNC(8);                     // in-order: tile kt resident
        __syncthreads();
        compute_tile(0);
        __syncthreads();                   // all waves done reading buf0

        if (kt + 2 < NT) {
            issue_tile(0);                 // tile kt+2 -> buf0
            WAIT_ASYNC(8);                 // tile kt+1 resident
        } else {
            WAIT_ASYNC(0);
        }
        __syncthreads();
        compute_tile(BUF_BYTES);
        __syncthreads();                   // all waves done reading buf1
    }

    // ---- epilogue: D layout -> VGPR r holds (M = r + 8*(lane>=16), N = lane&15)
    const int cm = (lane >> 4) * 8;
    const int cn = lane & 15;
    #pragma unroll
    for (int nt = 0; nt < 2; ++nt) {
        const int col = blockN + wn * 32 + nt * 16 + cn;
        const float bb = bias[col];
        #pragma unroll
        for (int mt = 0; mt < 4; ++mt) {
            const int rowb = blockM + wm * 64 + mt * 16 + cm;
            #pragma unroll
            for (int r = 0; r < 8; ++r)
                out[(size_t)(rowb + r) * OUT_F + col] = acc[mt][nt][r] + bb;
        }
    }
}

// ---------------------------------------------------------------------------
extern "C" void kernel_launch(void* const* d_in, const int* in_sizes, int n_in,
                              void* d_out, int out_size, void* d_ws, size_t ws_size,
                              hipStream_t stream) {
    const float* x        = (const float*)d_in[0];
    const float* ln_gamma = (const float*)d_in[1];
    const float* ln_beta  = (const float*)d_in[2];
    const float* spline_w = (const float*)d_in[3];
    const float* base_w   = (const float*)d_in[4];
    const float* base_b   = (const float*)d_in[5];
    float* out = (float*)d_out;

    __hip_bfloat16* Aext = (__hip_bfloat16*)d_ws;           // 8192*9216 bf16 ~151MB
    __hip_bfloat16* Wext = Aext + (size_t)NROWS * KTOT;     // 1024*9216 bf16 ~19MB

    prep_A_kernel<<<NROWS, 256, 0, stream>>>(x, ln_gamma, ln_beta, Aext);

    const int wthreads = (OUT_F * KTOT) / 4;
    prep_W_kernel<<<wthreads / 256, 256, 0, stream>>>(spline_w, base_w, Wext);

    dim3 grid(NROWS / 128, OUT_F / 128);
    gemm_kernel<<<grid, 256, LDS_TOTAL, stream>>>((const __bf16*)Aext,
                                                  (const __bf16*)Wext,
                                                  base_b, out);
}
